// MultiHeadAttention_9405978378694
// MI455X (gfx1250) — matrix operand order repaired and checked by reading, not compiled, not run on hardware
//
#include <hip/hip_runtime.h>
#include <hip/hip_bf16.h>

#define B_   2
#define S_   2048
#define DM_  2048
#define H_   16
#define DH_  128
#define DM3_ (3 * DM_)

typedef __attribute__((ext_vector_type(16))) _Float16 v16h;
typedef __attribute__((ext_vector_type(8)))  _Float16 v8h;
typedef __attribute__((ext_vector_type(8)))  float    v8f;

union V16U { v16h v; v8h h[2]; };

// ---------------------------------------------------------------------------
// Fragment loaders (wave32, ISA 7.12.2 layouts)
// A fragment: 16x32 f16, row m = lane%16, lane-half selects K+8 / K+24 groups.
__device__ __forceinline__ v16h load_a(const _Float16* base, int stride) {
  int lane = threadIdx.x & 31;
  int l = lane & 15, hh = lane >> 4;
  const _Float16* p = base + (size_t)l * stride + 8 * hh;
  V16U u;
  u.h[0] = *(const v8h*)(p);        // K = 8*hh + 0..7
  u.h[1] = *(const v8h*)(p + 16);   // K = 16 + 8*hh + 0..7
  return u.v;
}

// B fragment: 32x16 f16, column n = lane%16; lane-half selects K+8 / K+24
// groups (same K order as A). Column contiguous along K at `stride` halves.
__device__ __forceinline__ v16h load_b(const _Float16* base, int stride) {
  int lane = threadIdx.x & 31;
  int l = lane & 15, hh = lane >> 4;
  const _Float16* p = base + (size_t)l * stride + 8 * hh;
  V16U u;
  u.h[0] = *(const v8h*)(p);
  u.h[1] = *(const v8h*)(p + 16);
  return u.v;
}

__device__ __forceinline__ v8f wmma32(v16h a, v16h b, v8f c) {
  return __builtin_amdgcn_wmma_f32_16x16x32_f16(false, a, false, b, (short)0, c,
                                                false, false);
}

// ---------------------------------------------------------------------------
// Tensor Data Mover: generic 2D f16 tile -> LDS (tensor dims == tile dims).
#if __has_builtin(__builtin_amdgcn_tensor_load_to_lds)
#define USE_TDM 1
typedef __attribute__((ext_vector_type(4))) unsigned int u32x4;
typedef __attribute__((ext_vector_type(4))) int          i32x4;
typedef __attribute__((ext_vector_type(8))) int          i32x8;

// d0 contiguous f16 elements per row, `rows` rows, global row stride (elems).
// Lands in LDS densely packed: rows * d0 halves.
__device__ __forceinline__ void tdm_load_2d_f16(const _Float16* gsrc,
                                                unsigned lds_byte, unsigned d0,
                                                unsigned rows, unsigned row_stride) {
  unsigned long long ga = (unsigned long long)(size_t)gsrc;
  u32x4 g0;
  g0[0] = 1u;                                                 // count = 1
  g0[1] = lds_byte;                                           // lds_addr
  g0[2] = (unsigned)(ga & 0xffffffffu);                       // global_addr lo
  g0[3] = (unsigned)((ga >> 32) & 0x01ffffffu) | (2u << 30);  // addr hi | type=2
  i32x8 g1;
  g1[0] = (1 << 16);                          // wg_mask=0, data_size=1 (2B)
  g1[1] = (int)(d0 << 16);                    // tensor_dim0[15:0]
  g1[2] = (int)(rows << 16);                  // dim0 hi=0, tensor_dim1[15:0]
  g1[3] = (int)(d0 << 16);                    // dim1 hi=0, tile_dim0 = d0
  g1[4] = (int)rows;                          // tile_dim1 = rows, tile_dim2=0
  g1[5] = (int)row_stride;                    // tensor_dim0_stride[31:0]
  g1[6] = 0;
  g1[7] = 0;
  i32x4 z4 = {0, 0, 0, 0};
#if __clang_major__ >= 23
  i32x8 z8 = {0, 0, 0, 0, 0, 0, 0, 0};
  __builtin_amdgcn_tensor_load_to_lds(g0, g1, z4, z4, z8, 0);
#else
  __builtin_amdgcn_tensor_load_to_lds(g0, g1, z4, z4, 0);
#endif
}
#endif

// ---------------------------------------------------------------------------
// f32 -> f16 convert, 8 elements per thread
__global__ __launch_bounds__(256) void cvt_f16_kernel(const float* __restrict__ in,
                                                      _Float16* __restrict__ out,
                                                      int n8) {
  int i = blockIdx.x * blockDim.x + threadIdx.x;
  if (i >= n8) return;
  size_t base = (size_t)i * 8;
  float4 a = ((const float4*)(in + base))[0];
  float4 b = ((const float4*)(in + base))[1];
  v8h o;
  o[0] = (_Float16)a.x; o[1] = (_Float16)a.y; o[2] = (_Float16)a.z; o[3] = (_Float16)a.w;
  o[4] = (_Float16)b.x; o[5] = (_Float16)b.y; o[6] = (_Float16)b.z; o[7] = (_Float16)b.w;
  *(v8h*)(out + base) = o;
}

// ---------------------------------------------------------------------------
// Shared 32(M) x 64(N) wave-tile GEMM mainloop, ping-pong double buffered
// (two fragment buffer sets alternate roles -> no register copies).
__device__ __forceinline__ void loadset(const _Float16* const (&pa)[2],
                                        const _Float16* const (&pb)[4], int k0,
                                        v16h (&a)[2], v16h (&b)[4]) {
#pragma unroll
  for (int m = 0; m < 2; ++m) a[m] = load_a(pa[m] + k0, DM_);
#pragma unroll
  for (int n = 0; n < 4; ++n) b[n] = load_b(pb[n] + k0, DM_);
}

__device__ __forceinline__ void step8(v16h (&a)[2], v16h (&b)[4], v8f (&acc)[2][4]) {
#pragma unroll
  for (int m = 0; m < 2; ++m)
#pragma unroll
    for (int n = 0; n < 4; ++n) acc[m][n] = wmma32(a[m], b[n], acc[m][n]);
}

__device__ __forceinline__ void mainloop32x64(const _Float16* __restrict__ arow,
                                              const _Float16* __restrict__ brow,
                                              v8f (&acc)[2][4]) {
  const _Float16* pa[2] = {arow, arow + (size_t)16 * DM_};
  const _Float16* pb[4] = {brow, brow + (size_t)16 * DM_,
                           brow + (size_t)32 * DM_, brow + (size_t)48 * DM_};
  v16h a0[2], b0[4], a1[2], b1[4];
  loadset(pa, pb, 0, a0, b0);
  int k0 = 32;
#pragma unroll 1
  for (; k0 + 32 < DM_; k0 += 64) {
    loadset(pa, pb, k0, a1, b1);
    step8(a0, b0, acc);
    loadset(pa, pb, k0 + 32, a0, b0);
    step8(a1, b1, acc);
  }
  loadset(pa, pb, k0, a1, b1);   // k0 == DM_ - 32
  step8(a0, b0, acc);
  step8(a1, b1, acc);
}

// ---------------------------------------------------------------------------
// QKV GEMM epilogue scatters into q / k / vT head layouts.
__global__ __launch_bounds__(32) void gemm_qkv_kernel(
    const _Float16* __restrict__ xh, const _Float16* __restrict__ wh,
    const float* __restrict__ bias, _Float16* __restrict__ q,
    _Float16* __restrict__ k, _Float16* __restrict__ vt) {
  int n0 = blockIdx.x * 64;
  int m0 = blockIdx.y * 32;
  v8f acc[2][4] = {};
  mainloop32x64(xh + (size_t)m0 * DM_, wh + (size_t)n0 * DM_, acc);

  int lane = threadIdx.x & 31;
  int l = lane & 15, hh = lane >> 4;
  const float qscale = 0.08838834764831845f;  // DH^-0.5
#pragma unroll
  for (int m = 0; m < 2; ++m) {
#pragma unroll
    for (int n = 0; n < 4; ++n) {
#pragma unroll
      for (int r = 0; r < 8; ++r) {
        int row = m0 + 16 * m + r + 8 * hh;  // token index (b*S + s)
        int col = n0 + n * 16 + l;           // [0, 3*DM)
        float v = acc[m][n][r] + bias[col];
        int which = col >> 11;               // 0=q 1=k 2=v
        int cm = col & (DM_ - 1);
        int head = cm >> 7, d = cm & (DH_ - 1);
        int b = row >> 11, s = row & (S_ - 1);
        size_t bh = (size_t)(b * H_ + head);
        if (which == 0)
          q[(bh * S_ + s) * DH_ + d] = (_Float16)(v * qscale);
        else if (which == 1)
          k[(bh * S_ + s) * DH_ + d] = (_Float16)v;
        else
          vt[(bh * DH_ + d) * S_ + s] = (_Float16)v;
      }
    }
  }
}

// ---------------------------------------------------------------------------
// Rotary in-place on [B,H,S,DH] f16. Reference indexes fc tables by HEAD.
__global__ __launch_bounds__(256) void rotary_kernel(_Float16* __restrict__ buf,
                                                     const float* __restrict__ fcr,
                                                     const float* __restrict__ fci) {
  size_t i = (size_t)blockIdx.x * blockDim.x + threadIdx.x;
  const size_t total = (size_t)B_ * H_ * S_ * 64;
  if (i >= total) return;
  int d = (int)(i & 63);
  size_t row = i >> 6;                       // (b*H + h)*S + s
  int h = (int)((row >> 11) & (H_ - 1));
  float fr = fcr[h * 64 + d];
  float fi = fci[h * 64 + d];
  _Float16* p = buf + row * DH_;
  float re = (float)p[d];
  float im = (float)p[64 + d];
  p[d]      = (_Float16)(re * fr - im * fi);
  p[64 + d] = (_Float16)(re * fi + im * fr);
}

// ---------------------------------------------------------------------------
// Causal flash attention. Block = 4 waves = 64 query rows of one (b,h);
// all waves share double-buffered K/V tiles staged by the Tensor Data Mover
// (wave 0 drives the DMA; s_wait_tensorcnt retires tiles in order).
__global__ __launch_bounds__(128) void attn_kernel(
    const _Float16* __restrict__ q, const _Float16* __restrict__ k,
    const _Float16* __restrict__ vt, _Float16* __restrict__ ao) {
  __shared__ __align__(16) _Float16 plds[4][16 * 32];
#ifdef USE_TDM
  __shared__ __align__(16) _Float16 ktile[2][32 * DH_];   // [key][ch]
  __shared__ __align__(16) _Float16 vtile[2][DH_ * 32];   // [ch][key]
#endif
  int bh = blockIdx.y;                 // b*H + h
  int qbase = blockIdx.x * 64;
  int wid = threadIdx.x >> 5;
  int lane = threadIdx.x & 31;
  int l = lane & 15, hh = lane >> 4;
  int qrow0 = qbase + wid * 16;

  const _Float16* qbasep = q + ((size_t)bh * S_ + qrow0) * DH_;
  const _Float16* kb = k + (size_t)bh * S_ * DH_;
  const _Float16* vb = vt + (size_t)bh * DH_ * S_;

  v16h aq[4];
#pragma unroll
  for (int f = 0; f < 4; ++f) aq[f] = load_a(qbasep + 32 * f, DH_);

  v8f o[8] = {};
  float mrow[8], lrow[8];
#pragma unroll
  for (int r = 0; r < 8; ++r) { mrow[r] = -1e30f; lrow[r] = 0.0f; }

  int nt = qbase / 32 + 2;             // tiles of 32 keys, covers qbase+63

#ifdef USE_TDM
  unsigned koff[2] = {(unsigned)(size_t)&ktile[0][0], (unsigned)(size_t)&ktile[1][0]};
  unsigned voff[2] = {(unsigned)(size_t)&vtile[0][0], (unsigned)(size_t)&vtile[1][0]};
  if (wid == 0) {
    tdm_load_2d_f16(kb, koff[0], DH_, 32, DH_);
    tdm_load_2d_f16(vb, voff[0], 32, DH_, S_);
  }
#endif

#pragma unroll 1
  for (int i = 0; i < nt; ++i) {
    int t0 = i * 32;
#ifdef USE_TDM
    if (wid == 0) {
      if (i + 1 < nt) {
        int tn = t0 + 32;
        tdm_load_2d_f16(kb + (size_t)tn * DH_, koff[(i + 1) & 1], DH_, 32, DH_);
        tdm_load_2d_f16(vb + tn, voff[(i + 1) & 1], 32, DH_, S_);
        __builtin_amdgcn_s_wait_tensorcnt(2);   // retire current tile pair
      } else {
        __builtin_amdgcn_s_wait_tensorcnt(0);
      }
    }
    __syncthreads();                            // tile (i&1) visible to block
    const _Float16* kt  = &ktile[i & 1][0];
    const _Float16* vtl = &vtile[i & 1][0];
#endif

    v8f st0 = {}, st1 = {};
#ifdef USE_TDM
#pragma unroll
    for (int f = 0; f < 4; ++f)
      st0 = wmma32(aq[f], load_b(kt + 32 * f, DH_), st0);
#pragma unroll
    for (int f = 0; f < 4; ++f)
      st1 = wmma32(aq[f], load_b(kt + 16 * DH_ + 32 * f, DH_), st1);
#else
#pragma unroll
    for (int f = 0; f < 4; ++f)
      st0 = wmma32(aq[f], load_b(kb + (size_t)t0 * DH_ + 32 * f, DH_), st0);
#pragma unroll
    for (int f = 0; f < 4; ++f)
      st1 = wmma32(aq[f], load_b(kb + (size_t)(t0 + 16) * DH_ + 32 * f, DH_), st1);
#endif

#pragma unroll
    for (int r = 0; r < 8; ++r) {
      int row = qrow0 + r + 8 * hh;
      int c0 = t0 + l, c1 = c0 + 16;
      float s0 = (c0 <= row) ? st0[r] : -1e30f;
      float s1 = (c1 <= row) ? st1[r] : -1e30f;
      float mx = fmaxf(s0, s1);
      mx = fmaxf(mx, __shfl_xor(mx, 1));
      mx = fmaxf(mx, __shfl_xor(mx, 2));
      mx = fmaxf(mx, __shfl_xor(mx, 4));
      mx = fmaxf(mx, __shfl_xor(mx, 8));
      float mnew = fmaxf(mrow[r], mx);
      float alpha = __expf(mrow[r] - mnew);
      float p0 = __expf(s0 - mnew);
      float p1 = __expf(s1 - mnew);
      float ps = p0 + p1;
      ps += __shfl_xor(ps, 1);
      ps += __shfl_xor(ps, 2);
      ps += __shfl_xor(ps, 4);
      ps += __shfl_xor(ps, 8);
      lrow[r] = lrow[r] * alpha + ps;
      mrow[r] = mnew;
#pragma unroll
      for (int ct = 0; ct < 8; ++ct) o[ct][r] *= alpha;
      int rl = r + 8 * hh;
      plds[wid][rl * 32 + l]      = (_Float16)p0;
      plds[wid][rl * 32 + 16 + l] = (_Float16)p1;
    }
    // Wave-private P round trip: DS ops are in-order within a wave.
    v16h pa = load_a(&plds[wid][0], 32);
#pragma unroll
    for (int ct = 0; ct < 8; ++ct) {
#ifdef USE_TDM
      v16h bv = load_b(vtl + (size_t)(ct * 16) * 32, 32);
#else
      v16h bv = load_b(vb + (size_t)(ct * 16) * S_ + t0, S_);
#endif
      o[ct] = wmma32(pa, bv, o[ct]);
    }
#ifdef USE_TDM
    __syncthreads();   // all reads of tile (i&1) done before it is re-filled
#endif
  }

  int b = bh >> 4, head = bh & (H_ - 1);
#pragma unroll
  for (int r = 0; r < 8; ++r) {
    float inv = 1.0f / lrow[r];
    int s = qrow0 + r + 8 * hh;
    size_t orow = ((size_t)b * S_ + s) * DM_ + (size_t)head * DH_;
#pragma unroll
    for (int ct = 0; ct < 8; ++ct)
      ao[orow + ct * 16 + l] = (_Float16)(o[ct][r] * inv);
  }
}

// ---------------------------------------------------------------------------
// Output GEMM: out[m, n] = sum_k a[m,k] * w_out[n,k] + b_out[n]  (f32 out)
__global__ __launch_bounds__(32) void gemm_out_kernel(
    const _Float16* __restrict__ ah, const _Float16* __restrict__ wh,
    const float* __restrict__ bias, float* __restrict__ out) {
  int n0 = blockIdx.x * 64;
  int m0 = blockIdx.y * 32;
  v8f acc[2][4] = {};
  mainloop32x64(ah + (size_t)m0 * DM_, wh + (size_t)n0 * DM_, acc);

  int lane = threadIdx.x & 31;
  int l = lane & 15, hh = lane >> 4;
#pragma unroll
  for (int m = 0; m < 2; ++m) {
#pragma unroll
    for (int n = 0; n < 4; ++n) {
#pragma unroll
      for (int r = 0; r < 8; ++r) {
        int row = m0 + 16 * m + r + 8 * hh;
        int col = n0 + n * 16 + l;
        out[(size_t)row * DM_ + col] = acc[m][n][r] + bias[col];
      }
    }
  }
}

// ---------------------------------------------------------------------------
extern "C" void kernel_launch(void* const* d_in, const int* in_sizes, int n_in,
                              void* d_out, int out_size, void* d_ws, size_t ws_size,
                              hipStream_t stream) {
  const float* x      = (const float*)d_in[0];
  const float* w_qkv  = (const float*)d_in[1];
  const float* b_qkv  = (const float*)d_in[2];
  const float* w_out  = (const float*)d_in[3];
  const float* b_out  = (const float*)d_in[4];
  const float* fc_r   = (const float*)d_in[5];
  const float* fc_i   = (const float*)d_in[6];
  float* out = (float*)d_out;

  const size_t NX   = (size_t)B_ * S_ * DM_;       // 8,388,608
  const size_t NWQ  = (size_t)DM3_ * DM_;          // 12,582,912
  const size_t NWO  = (size_t)DM_ * DM_;           // 4,194,304
  const size_t NHED = (size_t)B_ * H_ * S_ * DH_;  // 8,388,608

  _Float16* xh    = (_Float16*)d_ws;
  _Float16* wqkvh = xh + NX;
  _Float16* wouth = wqkvh + NWQ;
  _Float16* qh    = wouth + NWO;
  _Float16* kh    = qh + NHED;
  _Float16* vth   = kh + NHED;
  _Float16* aoh   = vth + NHED;

  cvt_f16_kernel<<<(int)(NX / 8 / 256), 256, 0, stream>>>(x, xh, (int)(NX / 8));
  cvt_f16_kernel<<<(int)(NWQ / 8 / 256), 256, 0, stream>>>(w_qkv, wqkvh, (int)(NWQ / 8));
  cvt_f16_kernel<<<(int)(NWO / 8 / 256), 256, 0, stream>>>(w_out, wouth, (int)(NWO / 8));

  gemm_qkv_kernel<<<dim3(DM3_ / 64, (B_ * S_) / 32), 32, 0, stream>>>(
      xh, wqkvh, b_qkv, qh, kh, vth);

  const size_t rot_total = (size_t)B_ * H_ * S_ * 64;
  rotary_kernel<<<(int)(rot_total / 256), 256, 0, stream>>>(qh, fc_r, fc_i);
  rotary_kernel<<<(int)(rot_total / 256), 256, 0, stream>>>(kh, fc_r, fc_i);

  attn_kernel<<<dim3(S_ / 64, B_ * H_), 128, 0, stream>>>(qh, kh, vth, aoh);

  gemm_out_kernel<<<dim3(DM_ / 64, (B_ * S_) / 32), 32, 0, stream>>>(
      aoh, wouth, b_out, out);
}
